// Net_27814208209379
// MI455X (gfx1250) — matrix operand order repaired and checked by reading, not compiled, not run on hardware
//
#include <hip/hip_runtime.h>
#include <hip/hip_bf16.h>

typedef __bf16 bf16_t;
typedef bf16_t v16bf __attribute__((ext_vector_type(16)));
typedef bf16_t v8bf  __attribute__((ext_vector_type(8)));
typedef bf16_t v4bf  __attribute__((ext_vector_type(4)));
typedef float  v8f   __attribute__((ext_vector_type(8)));

#define NBINS 26        // 25 spline bins + root as bin 25
#define KSPL  5

// ---------------------------------------------------------------------------
// f32 -> bf16 convert (vectorized x4)
// ---------------------------------------------------------------------------
__global__ __launch_bounds__(256)
void cvt_bf16_kernel(const float* __restrict__ x, bf16_t* __restrict__ y, size_t n) {
    size_t i = ((size_t)blockIdx.x * 256 + threadIdx.x) * 4;
    if (i >= n) return;
    float4 v = *(const float4*)(x + i);
    v4bf o;
    o[0] = (bf16_t)v.x; o[1] = (bf16_t)v.y; o[2] = (bf16_t)v.z; o[3] = (bf16_t)v.w;
    *(v4bf*)(y + i) = o;
}

// ---------------------------------------------------------------------------
// Transpose + convert: W[bin][i][o] f32 -> WT[bin][o][i] bf16 (32x32 LDS tiles)
// so B fragments read contiguously along K (Cin).
// ---------------------------------------------------------------------------
__global__ __launch_bounds__(256)
void transpose_cvt_kernel(const float* __restrict__ W, bf16_t* __restrict__ WT, int Dm) {
    __shared__ float tile[32][33];
    const size_t binoff = (size_t)blockIdx.z * Dm * Dm;
    const int i0 = blockIdx.y * 32;
    const int o0 = blockIdx.x * 32;
    const int tx = threadIdx.x & 31;
    const int ty = threadIdx.x >> 5;        // 0..7
    const float* Wb = W + binoff;
    bf16_t* Tb = WT + binoff;
#pragma unroll
    for (int r = 0; r < 32; r += 8)
        tile[ty + r][tx] = Wb[(size_t)(i0 + ty + r) * Dm + (o0 + tx)];
    __syncthreads();
#pragma unroll
    for (int r = 0; r < 32; r += 8)
        Tb[(size_t)(o0 + ty + r) * Dm + (i0 + tx)] = (bf16_t)tile[tx][ty + r];
}

// ---------------------------------------------------------------------------
// Per-edge B-spline basis precompute (shared by both layers)
// ---------------------------------------------------------------------------
__global__ __launch_bounds__(256)
void edge_pre_kernel(const float* __restrict__ pts, const int* __restrict__ tails,
                     const int* __restrict__ heads, float4* __restrict__ basis4,
                     int4* __restrict__ wi4, int E) {
    int e = blockIdx.x * 256 + threadIdx.x;
    if (e >= E) return;
    const float2* p2 = (const float2*)pts;
    float2 pt = p2[tails[e]];
    float2 ph = p2[heads[e]];
    float px = fminf(fmaxf(0.5f * (pt.x - ph.x) * (1.0f / 256.0f) + 0.5f, 0.f), 1.f);
    float py = fminf(fmaxf(0.5f * (pt.y - ph.y) * (1.0f / 256.0f) + 0.5f, 0.f), 1.f);
    float vx = px * (KSPL - 1), vy = py * (KSPL - 1);
    float fx = fminf(floorf(vx), (float)(KSPL - 2));
    float fy = fminf(floorf(vy), (float)(KSPL - 2));
    float frx = vx - fx, fry = vy - fy;
    int ix = (int)fx, iy = (int)fy;
    float4 b; int4 w;
    b.x = (1.f - frx) * (1.f - fry); w.x = ix     + iy       * KSPL;
    b.y = frx         * (1.f - fry); w.y = (ix+1) + iy       * KSPL;
    b.z = (1.f - frx) * fry;         w.z = ix     + (iy + 1) * KSPL;
    b.w = frx         * fry;         w.w = (ix+1) + (iy + 1) * KSPL;
    basis4[e] = b;
    wi4[e]    = w;
}

// ---------------------------------------------------------------------------
// Fill f32 buffer (used to init agg = -inf)
// ---------------------------------------------------------------------------
__global__ __launch_bounds__(256)
void fill_kernel(float* __restrict__ p, float val, size_t n) {
    size_t i = ((size_t)blockIdx.x * 256 + threadIdx.x) * 4;
    if (i >= n) return;
    float4 v = {val, val, val, val};
    *(float4*)(p + i) = v;
}

// ---------------------------------------------------------------------------
// Batched GEMM over 26 weight bins: Z[m][bin][n] = (A[m][:] @ W_bin)[n]
// A: [M x K] bf16 row-major, WT: [bin][N][K] bf16 (W transposed), Z bf16.
// Block = 256 thr = 8 wave32 -> 128x128 C tile; wave = 32x64 = 2x4 WMMA tiles.
// ---------------------------------------------------------------------------
__global__ __launch_bounds__(256)
void gemm_bins_kernel(const bf16_t* __restrict__ A, const bf16_t* __restrict__ WT,
                      bf16_t* __restrict__ Z, int M, int Kd, int Nd) {
    const int lane = threadIdx.x & 31;
    const int wave = threadIdx.x >> 5;
    const int mw = wave >> 1;               // 0..3
    const int nw = wave & 1;                // 0..1
    const int bin = blockIdx.z;
    const int m0 = blockIdx.y * 128 + mw * 32;
    const int n0 = blockIdx.x * 128 + nw * 64;
    const bf16_t* Wb = WT + (size_t)bin * Kd * Nd;

    // A frag (16x32 bf16): lane<16 -> row=lane, K 0-15 ; lane>=16 -> K 16-31
    const int arow = lane & 15;
    const int akb  = (lane >> 4) << 4;      // 0 or 16
    // B frag (32x16 bf16): lane<16 -> col=lane, K 0-15 ; lane>=16 -> K 16-31
    const int bcol = lane & 15;
    const int bkb  = (lane >> 4) << 4;      // 0 or 16

    const v8f vzero = {0.f, 0.f, 0.f, 0.f, 0.f, 0.f, 0.f, 0.f};
    v8f acc[2][4];
#pragma unroll
    for (int i = 0; i < 2; ++i)
#pragma unroll
        for (int j = 0; j < 4; ++j) acc[i][j] = vzero;

    for (int k0 = 0; k0 < Kd; k0 += 32) {
        v16bf a[2], b[4];
#pragma unroll
        for (int i = 0; i < 2; ++i) {
            const bf16_t* p = A + (size_t)(m0 + 16 * i + arow) * Kd + k0 + akb;
            union { v16bf v; v8bf h[2]; } u;
            u.h[0] = *(const v8bf*)p;
            u.h[1] = *(const v8bf*)(p + 8);
            a[i] = u.v;
        }
#pragma unroll
        for (int j = 0; j < 4; ++j) {
            const bf16_t* p = Wb + (size_t)(n0 + 16 * j + bcol) * Kd + k0 + bkb;
            union { v16bf v; v8bf h[2]; } u;
            u.h[0] = *(const v8bf*)p;
            u.h[1] = *(const v8bf*)(p + 8);
            b[j] = u.v;
        }
#pragma unroll
        for (int i = 0; i < 2; ++i)
#pragma unroll
            for (int j = 0; j < 4; ++j)
                acc[i][j] = __builtin_amdgcn_wmma_f32_16x16x32_bf16(
                    false, a[i], false, b[j], (short)0, acc[i][j], false, false);
    }

    // C/D layout: VGPR r -> M = r + (lane>=16 ? 8 : 0), N = lane&15
    const int roff = (lane >> 4) << 3;
    const int ccol = lane & 15;
#pragma unroll
    for (int i = 0; i < 2; ++i)
#pragma unroll
        for (int j = 0; j < 4; ++j) {
#pragma unroll
            for (int r = 0; r < 8; ++r) {
                int row = m0 + 16 * i + roff + r;
                int col = n0 + 16 * j + ccol;
                Z[((size_t)row * NBINS + bin) * Nd + col] = (bf16_t)acc[i][j][r];
            }
        }
}

// ---------------------------------------------------------------------------
// Edge scatter-max: one block per edge; lane covers 4 channels.
// msg = sum_s basis_s * Z[tail][wi_s][c]; atomic fmax into agg[head][c].
// ---------------------------------------------------------------------------
__global__ __launch_bounds__(256)
void scatter_max_kernel(const bf16_t* __restrict__ Z, const float4* __restrict__ basis4,
                        const int4* __restrict__ wi4, const int* __restrict__ tails,
                        const int* __restrict__ heads, float* __restrict__ agg, int D) {
    const int e = blockIdx.x;
    const int c = threadIdx.x * 4;
    if (c >= D) return;
    const int t = tails[e];
    const int h = heads[e];
    const float4 bs = basis4[e];
    const int4   wi = wi4[e];
    const float bb[4] = {bs.x, bs.y, bs.z, bs.w};
    const int   ww[4] = {wi.x, wi.y, wi.z, wi.w};
    float m0 = 0.f, m1 = 0.f, m2 = 0.f, m3 = 0.f;
#pragma unroll
    for (int s = 0; s < 4; ++s) {
        const v4bf z = *(const v4bf*)(Z + ((size_t)t * NBINS + ww[s]) * D + c);
        m0 += bb[s] * (float)z[0];
        m1 += bb[s] * (float)z[1];
        m2 += bb[s] * (float)z[2];
        m3 += bb[s] * (float)z[3];
    }
    float* ap = agg + (size_t)h * D + c;
    // lowers to global_atomic_max_num_f32 (no return), device scope
    __hip_atomic_fetch_max(ap + 0, m0, __ATOMIC_RELAXED, __HIP_MEMORY_SCOPE_AGENT);
    __hip_atomic_fetch_max(ap + 1, m1, __ATOMIC_RELAXED, __HIP_MEMORY_SCOPE_AGENT);
    __hip_atomic_fetch_max(ap + 2, m2, __ATOMIC_RELAXED, __HIP_MEMORY_SCOPE_AGENT);
    __hip_atomic_fetch_max(ap + 3, m3, __ATOMIC_RELAXED, __HIP_MEMORY_SCOPE_AGENT);
}

// ---------------------------------------------------------------------------
// Layer-1 epilogue: h1 = relu( finite(agg) + Z[:,root,:] + bias ) -> bf16
// ---------------------------------------------------------------------------
__global__ __launch_bounds__(256)
void finalize1_kernel(const float* __restrict__ agg, const bf16_t* __restrict__ Z,
                      const float* __restrict__ bias, bf16_t* __restrict__ h1bf,
                      int D, size_t total) {
    size_t i = (size_t)blockIdx.x * 256 + threadIdx.x;
    if (i >= total) return;
    size_t n = i / D;
    int    c = (int)(i - n * D);
    float a = agg[i];
    if (!__builtin_isfinite(a)) a = 0.f;
    float r = a + (float)Z[(n * NBINS + 25) * D + c] + bias[c];
    h1bf[i] = (bf16_t)fmaxf(r, 0.f);
}

// ---------------------------------------------------------------------------
// Layer-2 epilogue: out = descs + 0.1 * ( finite(agg) + Z[:,root,:] + bias2 )
// ---------------------------------------------------------------------------
__global__ __launch_bounds__(256)
void finalize2_kernel(const float* __restrict__ descs, const float* __restrict__ agg,
                      const bf16_t* __restrict__ Z, const float* __restrict__ bias,
                      float* __restrict__ out, int D, size_t total) {
    size_t i = (size_t)blockIdx.x * 256 + threadIdx.x;
    if (i >= total) return;
    size_t n = i / D;
    int    c = (int)(i - n * D);
    float a = agg[i];
    if (!__builtin_isfinite(a)) a = 0.f;
    float r = a + (float)Z[(n * NBINS + 25) * D + c] + bias[c];
    out[i] = descs[i] + 0.1f * r;
}

// ---------------------------------------------------------------------------
extern "C" void kernel_launch(void* const* d_in, const int* in_sizes, int n_in,
                              void* d_out, int out_size, void* d_ws, size_t ws_size,
                              hipStream_t stream) {
    const float* descs = (const float*)d_in[0];
    const float* pts   = (const float*)d_in[1];
    const float* W1    = (const float*)d_in[2];
    const float* root1 = (const float*)d_in[3];
    const float* bias1 = (const float*)d_in[4];
    const float* W2    = (const float*)d_in[5];
    const float* root2 = (const float*)d_in[6];
    const float* bias2 = (const float*)d_in[7];
    const int*   tails = (const int*)d_in[8];
    const int*   heads = (const int*)d_in[9];
    float* out = (float*)d_out;

    const int    D  = in_sizes[4];             // 1024
    const size_t N  = (size_t)in_sizes[0] / D; // 4096
    const int    E  = in_sizes[8];             // 65536
    const size_t ND = N * (size_t)D;

    // Carve workspace (~300 MB total)
    char* base = (char*)d_ws;
    size_t off = 0;
    auto alloc = [&](size_t bytes) {
        void* p = base + off;
        off = (off + bytes + 255) & ~(size_t)255;
        return p;
    };
    bf16_t* xbf    = (bf16_t*)alloc(ND * 2);
    bf16_t* h1bf   = (bf16_t*)alloc(ND * 2);
    float4* basis4 = (float4*)alloc((size_t)E * 16);
    int4*   wi4    = (int4*)  alloc((size_t)E * 16);
    float*  agg    = (float*) alloc(ND * 4);
    bf16_t* WT     = (bf16_t*)alloc((size_t)NBINS * D * D * 2);
    bf16_t* Z      = (bf16_t*)alloc(ND * (size_t)NBINS * 2);
    (void)ws_size; (void)n_in; (void)out_size;

    const dim3 b256(256);
    const int cvtBlocks = (int)((ND / 4 + 255) / 256);
    const dim3 tgridW(D / 32, D / 32, 25);
    const dim3 tgridR(D / 32, D / 32, 1);
    const dim3 ggrid(D / 128, (unsigned)(N / 128), NBINS);
    const int finBlocks = (int)((ND + 255) / 256);
    const float ninf = -__builtin_inff();

    // Shared precompute
    edge_pre_kernel<<<(E + 255) / 256, b256, 0, stream>>>(pts, tails, heads, basis4, wi4, E);
    cvt_bf16_kernel<<<cvtBlocks, b256, 0, stream>>>(descs, xbf, ND);

    // ----- Layer 1 -----
    transpose_cvt_kernel<<<tgridW, b256, 0, stream>>>(W1, WT, D);
    transpose_cvt_kernel<<<tgridR, b256, 0, stream>>>(root1, WT + (size_t)25 * D * D, D);
    fill_kernel<<<cvtBlocks, b256, 0, stream>>>(agg, ninf, ND);
    gemm_bins_kernel<<<ggrid, b256, 0, stream>>>(xbf, WT, Z, (int)N, D, D);
    scatter_max_kernel<<<E, b256, 0, stream>>>(Z, basis4, wi4, tails, heads, agg, D);
    finalize1_kernel<<<finBlocks, b256, 0, stream>>>(agg, Z, bias1, h1bf, D, ND);

    // ----- Layer 2 -----
    transpose_cvt_kernel<<<tgridW, b256, 0, stream>>>(W2, WT, D);
    transpose_cvt_kernel<<<tgridR, b256, 0, stream>>>(root2, WT + (size_t)25 * D * D, D);
    fill_kernel<<<cvtBlocks, b256, 0, stream>>>(agg, ninf, ND);
    gemm_bins_kernel<<<ggrid, b256, 0, stream>>>(h1bf, WT, Z, (int)N, D, D);
    scatter_max_kernel<<<E, b256, 0, stream>>>(Z, basis4, wi4, tails, heads, agg, D);
    finalize2_kernel<<<finBlocks, b256, 0, stream>>>(descs, agg, Z, bias2, out, D, ND);
}
